// MyModel1_58935541236264
// MI455X (gfx1250) — matrix operand order repaired and checked by reading, not compiled, not run on hardware
//
#include <hip/hip_runtime.h>

typedef __attribute__((ext_vector_type(16))) __bf16 v16bf;
typedef __attribute__((ext_vector_type(8)))  float  v8f;
typedef __attribute__((ext_vector_type(4)))  float  v4f;

typedef __attribute__((ext_vector_type(4))) unsigned int u32x4;
typedef __attribute__((ext_vector_type(8))) int          i32x8;
typedef __attribute__((ext_vector_type(4))) int          i32x4;

#define NN   8192   // nodes
#define FIN  200
#define FOUT 100
#define KP   224    // padded feature dim (14 * 16)
#define NPO  112    // padded output dim (7 * 16)

#define LROW 72     // LDS row stride in bf16 elems: 64 data + 8 pad (144 B)
#define LBUF (KP * LROW)   // 16128 elems = 32256 B per buffer

// ---------------- column sums of A (for degree) ----------------
__global__ void colsum_part(const float* __restrict__ A, float* __restrict__ partial) {
    int j  = blockIdx.x * 256 + threadIdx.x;      // column
    int r0 = blockIdx.y * 1024;                   // row chunk
    const float* p = A + (size_t)r0 * NN + j;
    float s = 0.0f;
    #pragma unroll 8
    for (int i = 0; i < 1024; ++i) s += p[(size_t)i * NN];
    partial[blockIdx.y * NN + j] = s;
}

__global__ void colsum_fin(const float* __restrict__ partial, float* __restrict__ dinv) {
    int j = blockIdx.x * 256 + threadIdx.x;
    float s = 1.0f;                               // +I diagonal
    #pragma unroll
    for (int c = 0; c < 8; ++c) s += partial[c * NN + j];
    dinv[j] = rsqrtf(s);
}

// ---------------- build bf16 operands + G init ----------------
__global__ void prep(const float* __restrict__ h0, const float* __restrict__ W,
                     const float* __restrict__ dinv,
                     __bf16* __restrict__ h0sT, __bf16* __restrict__ Wp,
                     float* __restrict__ G) {
    int idx = blockIdx.x * 256 + threadIdx.x;
    const int n1 = KP * NN;
    const int n2 = NPO * KP;
    const int n3 = NN * KP;
    if (idx < n1) {
        int f = idx / NN, j = idx % NN;
        float v = (f < FIN) ? dinv[j] * h0[j * FIN + f] : 0.0f;
        h0sT[idx] = (__bf16)v;
    } else if (idx < n1 + n2) {
        int k2 = idx - n1;
        int o = k2 / KP, f = k2 % KP;
        float v = (o < FOUT && f < FIN) ? W[o * FIN + f] : 0.0f;
        Wp[k2] = (__bf16)v;
    } else if (idx < n1 + n2 + n3) {
        int k3 = idx - n1 - n2;
        int i = k3 / KP, f = k3 % KP;
        float v = (f < FIN) ? dinv[i] * h0[i * FIN + f] : 0.0f;
        G[k3] = v;
    }
}

// ---------------- TDM issue: DMA a 224 x 64 bf16 slice of h0sT into LDS ------
// D# per CDNA5 ISA ch.8: group0 = {count/type/lds_addr/global_addr},
// group1 = {data_size, pad, tensor dims/strides, tile dims}, groups2/3 = 0 (2D).
// Pad: 4 DWORDs (16B) after every 128B row -> 144B LDS row stride, which makes
// the per-lane ds_load_b128 fragment reads bank-conflict-free (stride 36 dwords).
__device__ __forceinline__ void tdm_issue(const __bf16* gsrc, unsigned lds_off) {
    unsigned long long ga = (unsigned long long)(size_t)gsrc;
    u32x4 g0;
    g0.x = 1u;                                          // count=1 (valid user D#)
    g0.y = lds_off;                                     // LDS byte address
    g0.z = (unsigned)(ga & 0xffffffffu);                // global_addr[31:0]
    g0.w = (unsigned)((ga >> 32) & 0x01ffffffu)         // global_addr[56:32]
         | (2u << 30);                                  // type=2 ("image")
    i32x8 g1;
    g1[0] = (1 << 16)            // data_size = 2 bytes
          | (1 << 20)            // pad_enable
          | (4 << 22)            // pad_interval: every 32 DWORDs (128B)
          | (3 << 25);           // pad_amount: 4 DWORDs (16B)   [encoding: val+1]
    g1[1] = (int)(8192u << 16);  // [63:48] tensor_dim0 = 8192 (lo16)
    g1[2] = (int)(224u << 16);   // [79:64] dim0 hi = 0 | [95:80] tensor_dim1 = 224
    g1[3] = (int)(64u << 16);    // [111:96] dim1 hi = 0 | [127:112] tile_dim0 = 64
    g1[4] = 224;                 // [143:128] tile_dim1 = 224 | tile_dim2 = 0
    g1[5] = 8192;                // tensor_dim0_stride lo32 (elements)
    g1[6] = 0;                   // stride hi | tensor_dim1_stride lo
    g1[7] = 0;
    i32x4 g2 = {};
    i32x4 g3 = {};
#if __clang_major__ >= 23
    i32x8 g4 = {};
    __builtin_amdgcn_tensor_load_to_lds(g0, g1, g2, g3, g4, 0);
#else
    __builtin_amdgcn_tensor_load_to_lds(g0, g1, g2, g3, 0);
#endif
}

// ---------------- main GEMM helpers ----------------
__device__ __forceinline__ void load_achunk(v4f a[8], const float* __restrict__ aP, int k) {
    a[0] = *(const v4f*)(aP + k);
    a[1] = *(const v4f*)(aP + k + 4);
    a[2] = *(const v4f*)(aP + k + 8);
    a[3] = *(const v4f*)(aP + k + 12);
    a[4] = *(const v4f*)(aP + k + 32);
    a[5] = *(const v4f*)(aP + k + 36);
    a[6] = *(const v4f*)(aP + k + 40);
    a[7] = *(const v4f*)(aP + k + 44);
}

// Two 32-K WMMA steps against one staged LDS buffer (lb = per-lane base).
__device__ __forceinline__ void compute_chunk(v8f acc[7], const v4f a[8],
                                              const __bf16* lb) {
    #pragma unroll
    for (int s = 0; s < 2; ++s) {
        v16bf bf[7];
        #pragma unroll
        for (int t = 0; t < 7; ++t)
            bf[t] = *(const v16bf*)(lb + t * (16 * LROW) + s * 32);
        v16bf af;
        #pragma unroll
        for (int e = 0; e < 4; ++e) {
            af[e]      = (__bf16)a[s * 4 + 0][e];
            af[e + 4]  = (__bf16)a[s * 4 + 1][e];
            af[e + 8]  = (__bf16)a[s * 4 + 2][e];
            af[e + 12] = (__bf16)a[s * 4 + 3][e];
        }
        #pragma unroll
        for (int t = 0; t < 7; ++t)
            acc[t] = __builtin_amdgcn_wmma_f32_16x16x32_bf16(
                false, af, false, bf[t], (short)0, acc[t], false, false);
    }
}

// ---------------- main GEMM: G = dinv * (A @ h0s + h0s) ----------------
// Block: 256 threads = 8 waves, 64 rows of A, all 14 N-tiles (7 per wave-half).
// TDM double-buffers 224x64 slices of h0sT in LDS; A streams from HBM.
__global__ void __launch_bounds__(256) gemm1(const float* __restrict__ A,
                                             const __bf16* __restrict__ h0sT,
                                             const float* __restrict__ dinv,
                                             float* __restrict__ G) {
    __shared__ __bf16 smem[2][LBUF];

    const int lane = threadIdx.x & 31;
    const int wave = threadIdx.x >> 5;
    const int lo = lane & 15, hi = lane >> 4;
    const int pair = wave >> 1, half = wave & 1;
    const int mBase = blockIdx.x * 64 + pair * 16;
    const int row = mBase + lo;

    v8f acc[7] = {};

    const float* aPtr = A + (size_t)row * NN + hi * 16;
    const int    frow = half * NPO + lo;                 // 0..223: h0sT row
    const __bf16* lb0 = &smem[0][frow * LROW + hi * 16]; // per-lane frag base
    const __bf16* lb1 = &smem[1][frow * LROW + hi * 16];
    const unsigned ldsOff0 = (unsigned)(size_t)&smem[0][0];
    const unsigned ldsOff1 = (unsigned)(size_t)&smem[1][0];

    // Prologue: stage first B slice, preload first A chunk.
    v4f c0[8], c1[8];
    load_achunk(c0, aPtr, 0);
    if (wave == 0) {                       // uniform branch: tensor ops ignore EXEC,
        tdm_issue(h0sT, ldsOff0);          // so only one wave may reach the instr.
        __builtin_amdgcn_s_wait_tensorcnt(0);
    }
    __syncthreads();

    for (int kb = 0; kb < NN; kb += 128) {
        // ---- stage A: compute [kb, kb+64) from buf0; fill buf1 with [kb+64, kb+128)
        if (wave == 0) tdm_issue(h0sT + kb + 64, ldsOff1);
        load_achunk(c1, aPtr, kb + 64);
        if (kb + 1088 <= NN) __builtin_prefetch((const void*)(aPtr + kb + 1024), 0, 1);
        compute_chunk(acc, c0, lb0);
        if (wave == 0) __builtin_amdgcn_s_wait_tensorcnt(0);
        __syncthreads();

        // ---- stage B: compute [kb+64, kb+128) from buf1; fill buf0 with next
        if (wave == 0 && kb + 128 < NN) tdm_issue(h0sT + kb + 128, ldsOff0);
        if (kb + 128 < NN) load_achunk(c0, aPtr, kb + 128);
        compute_chunk(acc, c1, lb1);
        if (wave == 0) __builtin_amdgcn_s_wait_tensorcnt(0);
        __syncthreads();
    }

    // Epilogue: G[i][f] = dinv[i] * (acc + G_init[i][f])   (coalesced RMW)
    v4f dv0 = *(const v4f*)(dinv + mBase + hi * 8);
    v4f dv1 = *(const v4f*)(dinv + mBase + hi * 8 + 4);
    #pragma unroll
    for (int t = 0; t < 7; ++t) {
        const int f = (half * 7 + t) * 16 + lo;
        #pragma unroll
        for (int r = 0; r < 8; ++r) {
            const int i = mBase + hi * 8 + r;       // C layout: M = r + 8*hi
            float dvi = (r < 4) ? dv0[r] : dv1[r - 4];
            float* gp = G + (size_t)i * KP + f;
            *gp = dvi * (acc[t][r] + *gp);
        }
    }
}

// ---------------- second GEMM: out = relu(G @ Wp^T + b) ----------------
__global__ void __launch_bounds__(256) gemm2(const float* __restrict__ G,
                                             const __bf16* __restrict__ Wp,
                                             const float* __restrict__ bias,
                                             float* __restrict__ out) {
    const int lane = threadIdx.x & 31;
    const int wave = threadIdx.x >> 5;
    const int lo = lane & 15, hi = lane >> 4;
    const int mBase = blockIdx.x * 128 + wave * 16;
    const int row = mBase + lo;

    v8f acc[7] = {};

    const float*  aPtr = G  + (size_t)row * KP + hi * 16;
    const __bf16* bPtr = Wp + (size_t)lo * KP + hi * 16;

    #pragma unroll
    for (int k = 0; k < KP; k += 32) {
        v4f a0 = *(const v4f*)(aPtr + k);
        v4f a1 = *(const v4f*)(aPtr + k + 4);
        v4f a2 = *(const v4f*)(aPtr + k + 8);
        v4f a3 = *(const v4f*)(aPtr + k + 12);
        v16bf af;
        #pragma unroll
        for (int e = 0; e < 4; ++e) {
            af[e]      = (__bf16)a0[e];
            af[e + 4]  = (__bf16)a1[e];
            af[e + 8]  = (__bf16)a2[e];
            af[e + 12] = (__bf16)a3[e];
        }
        #pragma unroll
        for (int t = 0; t < 7; ++t) {
            v16bf bf = *(const v16bf*)(bPtr + (size_t)t * 16 * KP + k);
            acc[t] = __builtin_amdgcn_wmma_f32_16x16x32_bf16(
                false, af, false, bf, (short)0, acc[t], false, false);
        }
    }

    #pragma unroll
    for (int t = 0; t < 7; ++t) {
        const int o = t * 16 + lo;
        if (o < FOUT) {
            float bo = bias[o];
            #pragma unroll
            for (int r = 0; r < 8; ++r) {
                const int i = mBase + hi * 8 + r;
                float v = acc[t][r] + bo;
                out[(size_t)i * FOUT + o] = v > 0.0f ? v : 0.0f;
            }
        }
    }
}

extern "C" void kernel_launch(void* const* d_in, const int* in_sizes, int n_in,
                              void* d_out, int out_size, void* d_ws, size_t ws_size,
                              hipStream_t stream) {
    const float* A  = (const float*)d_in[0];   // [8192,8192]
    const float* h0 = (const float*)d_in[1];   // [8192,200]
    const float* W  = (const float*)d_in[2];   // [100,200]
    const float* b  = (const float*)d_in[3];   // [100]
    float* out = (float*)d_out;                // [8192,100]

    char* ws = (char*)d_ws;
    float*  dinv    = (float*)(ws);                          //    32768 B
    float*  partial = (float*)(ws + 32768);                  //   262144 B
    __bf16* h0sT    = (__bf16*)(ws + 294912);                //  3670016 B
    __bf16* Wp      = (__bf16*)(ws + 294912 + 3670016);      //    50176 B
    float*  G       = (float*)(ws + 4018176);                //  7340032 B (4KB aligned)

    colsum_part<<<dim3(32, 8), 256, 0, stream>>>(A, partial);
    colsum_fin<<<32, 256, 0, stream>>>(partial, dinv);

    const int prepTotal = KP * NN + NPO * KP + NN * KP;      // 3,695,104
    prep<<<(prepTotal + 255) / 256, 256, 0, stream>>>(h0, W, dinv, h0sT, Wp, G);

    gemm1<<<128, 256, 0, stream>>>(A, h0sT, dinv, G);
    gemm2<<<64, 256, 0, stream>>>(G, Wp, b, out);
}
